// VectorQuantizer_80290118632195
// MI455X (gfx1250) — matrix operand order repaired and checked by reading, not compiled, not run on hardware
//
#include <hip/hip_runtime.h>
#include <hip/hip_bf16.h>

// ---------------------------------------------------------------------------
// VQ-VAE vector quantizer for MI455X (gfx1250, wave32, WMMA + async-to-LDS).
// B=32, D=256, T=1024 -> N=32768 flat vectors; K=8192 codewords.
// Distance argmin via bf16 WMMA (fp32 accum): argmax_k (x.w_k - 0.5*||w_k||^2).
// bf16 is safe here (codewords ~1e-4 magnitude need f32 exponent range; fp8
// e4m3 would flush them to zero). ||w||^2/2 and loss stay fp32.
// ---------------------------------------------------------------------------

#define NKD_B   32
#define NKD_D   256
#define NKD_T   1024
#define NKD_N   (NKD_B * NKD_T)      // 32768
#define NKD_K   8192
#define OUT_Q_OFF   1                // d_out: [loss][quantized 8388608][indices 32768]
#define OUT_IDX_OFF (1 + NKD_B * NKD_D * NKD_T)

typedef __attribute__((ext_vector_type(16))) __bf16 v16bf;
typedef __attribute__((ext_vector_type(8)))  float  v8f;
typedef __attribute__((ext_vector_type(4)))  int    v4i;

typedef __attribute__((address_space(1))) v4i* as1_v4i_ptr;
typedef __attribute__((address_space(3))) v4i* as3_v4i_ptr;

union BFrag {
    unsigned int u[8];
    uint4        q[2];
    v16bf        v;
};

__device__ __forceinline__ unsigned int f2bf_rn(float f) {
    unsigned int u = __float_as_uint(f);
    return (u + 0x7FFFu + ((u >> 16) & 1u)) >> 16;   // RNE to bf16
}
__device__ __forceinline__ unsigned int pack_bf2(float lo, float hi) {
    return f2bf_rn(lo) | (f2bf_rn(hi) << 16);
}

// ---- CDNA5 async global->LDS copy (ASYNCcnt-tracked DMA, no VGPR bounce) ----
// Builtin signature wants (v4i AS1*, v4i AS3*, Imm offset, Imm cpol).
// Cast through integers: AS1 is the full 64-bit address; generic shared
// addresses carry the LDS offset in addr[31:0] (flat aperture rule).
__device__ __forceinline__ void async_copy_b128(const void* g, void* l) {
#if __has_builtin(__builtin_amdgcn_global_load_async_to_lds_b128)
    __builtin_amdgcn_global_load_async_to_lds_b128(
        (as1_v4i_ptr)(unsigned long long)g,
        (as3_v4i_ptr)(unsigned int)(unsigned long long)l,
        0, 0);
#else
    asm volatile("global_load_async_to_lds_b128 %0, %1, off"
                 :: "v"((unsigned int)(unsigned long long)l),
                    "v"((unsigned long long)g)
                 : "memory");
#endif
}

__device__ __forceinline__ void wait_async0() {
#if __has_builtin(__builtin_amdgcn_s_wait_asynccnt)
    __builtin_amdgcn_s_wait_asynccnt(0);
#else
    asm volatile("s_wait_asynccnt 0x0" ::: "memory");
#endif
}

// ---------------------------------------------------------------------------
// Kernel 1: wnh[k] = 0.5*||w_k||^2 (fp32), w -> bf16, zero loss accumulator.
// ---------------------------------------------------------------------------
__global__ __launch_bounds__(256)
void vq_prep(const float* __restrict__ w,
             float* __restrict__ wnh,
             unsigned short* __restrict__ wbf,
             float* __restrict__ out) {
    const int k   = blockIdx.x;
    const int tid = threadIdx.x;           // == d
    __shared__ float red[256];

    float v = w[(size_t)k * NKD_D + tid];
    wbf[(size_t)k * NKD_D + tid] = (unsigned short)f2bf_rn(v);

    red[tid] = v * v;
    __syncthreads();
    #pragma unroll
    for (int s = 128; s > 0; s >>= 1) {
        if (tid < s) red[tid] += red[tid + s];
        __syncthreads();
    }
    if (tid == 0) wnh[k] = 0.5f * red[0];
    if (k == 0 && tid == 0) out[0] = 0.0f;   // loss accumulator
}

// ---------------------------------------------------------------------------
// Kernel 2: argmin via v_wmma_f32_16x16x32_bf16 with double-buffered
// GLOBAL_LOAD_ASYNC_TO_LDS staging of the codebook.
//   A (16x32 bf16) = codeword tile from LDS (swept over K).
//   B (32x16 bf16) = 16 x-rows per wave, in registers the whole sweep.
//   C (16x16 f32)  = scores; one x-column per lane -> 2-reg running argmax.
// Grid: 256 blocks x 256 threads (8 waves); block owns 128 x-rows.
// ---------------------------------------------------------------------------
__global__ __launch_bounds__(256)
void vq_argmin(const float* __restrict__ x,            // [B, D, T]
               const unsigned short* __restrict__ wbf, // [K, D] bf16
               const float* __restrict__ wnh,          // [K]
               int* __restrict__ idxbuf,               // [N]
               float* __restrict__ out) {
    __shared__ uint4 wt4[2 * 2048];   // double buffer: 2 x (64 codewords x 256 bf16) = 64 KB
    __shared__ float wnh_s[2 * 64];

    const int tid  = threadIdx.x;
    const int lane = tid & 31;
    const int wv   = tid >> 5;           // wave id 0..7
    const int nloc = lane & 15;
    const int grp  = lane >> 4;          // lane half

    const int n0 = blockIdx.x * 128;
    const int b  = n0 >> 10;
    const int t0 = n0 & 1023;
    const int t  = t0 + wv * 16 + nloc;  // this lane's x-row (time index)

    // ---- load this wave's 16 x-rows as WMMA B fragments (bf16, in regs) ----
    // B (32x16) in contiguous k order, not AMD's table: lane = n + 16*(k>=16); VGPR j holds k=16*grp+2j,+1.
    BFrag xb[8];
    const float* xbase = x + (size_t)b * NKD_D * NKD_T + t;
    #pragma unroll
    for (int c = 0; c < 8; ++c) {
        #pragma unroll
        for (int j = 0; j < 8; ++j) {
            const int d = c * 32 + grp * 16 + 2 * j;
            const float f0 = xbase[(size_t)d * NKD_T];
            const float f1 = xbase[(size_t)(d + 1) * NKD_T];
            xb[c].u[j] = pack_bf2(f0, f1);
        }
    }

    const uint4* __restrict__ wbf4 = (const uint4*)wbf;

    // ---- prologue: async-stage chunk 0 into buffer 0 ----
    #pragma unroll
    for (int j = 0; j < 8; ++j)
        async_copy_b128(wbf4 + tid + 256 * j, &wt4[tid + 256 * j]);
    if (tid < 64) wnh_s[tid] = wnh[tid];
    wait_async0();
    __syncthreads();

    float bestv = -3.402823466e+38f;
    int   besti = 0;

    for (int kc = 0; kc < NKD_K; kc += 64) {
        const int cur = (kc >> 6) & 1;
        const uint4* wt = wt4 + cur * 2048;
        const float* wn = wnh_s + cur * 64;

        // ---- async-prefetch next chunk into the other buffer ----
        if (kc + 64 < NKD_K) {
            const uint4* src = wbf4 + (size_t)(kc + 64) * 32;
            uint4*       dst = wt4 + (cur ^ 1) * 2048;
            #pragma unroll
            for (int j = 0; j < 8; ++j)
                async_copy_b128(src + tid + 256 * j, dst + tid + 256 * j);
            if (tid < 64) wnh_s[(cur ^ 1) * 64 + tid] = wnh[kc + 64 + tid];
        }

        // ---- compute on current buffer: 4 subtiles x 8 k-steps ----
        #pragma unroll
        for (int sub = 0; sub < 4; ++sub) {
            v8f acc = {};
            #pragma unroll
            for (int c = 0; c < 8; ++c) {
                // A (16x32) in this code's contiguous k order, matching B: row m = lane&15;
                // lanes 0-15 hold k {0..15}, lanes 16-31 hold {16..31}.
                BFrag a;
                const int rbase = (sub * 16 + nloc) * 32 + c * 4 + 2 * grp;
                a.q[0] = wt[rbase];
                a.q[1] = wt[rbase + 1];
                acc = __builtin_amdgcn_wmma_f32_16x16x32_bf16(
                        false, a.v, false, xb[c].v, (short)0, acc, false, false);
            }
            // C/D layout: VGPR v holds (m = v + 8*grp, n = lane&15).
            #pragma unroll
            for (int v = 0; v < 8; ++v) {
                const int ml = sub * 16 + grp * 8 + v;
                const float s = acc[v] - wn[ml];
                if (s > bestv) { bestv = s; besti = kc + ml; }  // ascending k: first-min tie-break
            }
        }

        wait_async0();      // next buffer's DMA complete (no-op on last iter)
        __syncthreads();    // all waves done with current buffer + staging visible
    }

    // merge the two lane halves covering the same x-column
    const float ov = __shfl_xor(bestv, 16);
    const int   oi = __shfl_xor(besti, 16);
    if (ov > bestv || (ov == bestv && oi < besti)) { bestv = ov; besti = oi; }

    if (lane < 16) {
        const int nflat = n0 + wv * 16 + nloc;
        idxbuf[nflat] = besti;
        out[(size_t)OUT_IDX_OFF + nflat] = (float)besti;
    }
}

// ---------------------------------------------------------------------------
// Kernel 3: quantized_out[b][d][t] = w[idx[n]][d] (coalesced stores over t),
// and loss = 1.25 * mean((q - x)^2) via block reduction + one atomicAdd.
// ---------------------------------------------------------------------------
__global__ __launch_bounds__(256)
void vq_out(const float* __restrict__ x,
            const float* __restrict__ w,
            const int* __restrict__ idxbuf,
            float* __restrict__ out) {
    __shared__ int   sidx[128];
    __shared__ float red[256];

    const int tid = threadIdx.x;
    const int n0  = blockIdx.x * 128;
    const int b   = n0 >> 10;
    const int t0  = n0 & 1023;

    if (tid < 128) sidx[tid] = idxbuf[n0 + tid];
    __syncthreads();

    const int tl = tid & 127;
    const int d0 = tid >> 7;
    const int t  = t0 + tl;
    const float* wrow = w + (size_t)sidx[tl] * NKD_D;

    float lsum = 0.0f;
    #pragma unroll 4
    for (int d = d0; d < NKD_D; d += 2) {
        const size_t off = ((size_t)(b * NKD_D + d) << 10) + t;
        const float q  = wrow[d];
        const float xv = x[off];
        out[OUT_Q_OFF + off] = q;
        const float df = q - xv;
        lsum += df * df;
    }

    red[tid] = lsum;
    __syncthreads();
    #pragma unroll
    for (int s = 128; s > 0; s >>= 1) {
        if (tid < s) red[tid] += red[tid + s];
        __syncthreads();
    }
    if (tid == 0)
        atomicAdd(out, red[0] * (1.25f / (float)(NKD_N * NKD_D)));
}

// ---------------------------------------------------------------------------
extern "C" void kernel_launch(void* const* d_in, const int* in_sizes, int n_in,
                              void* d_out, int out_size, void* d_ws, size_t ws_size,
                              hipStream_t stream) {
    (void)in_sizes; (void)n_in; (void)out_size; (void)ws_size;

    const float* x = (const float*)d_in[0];   // [32, 256, 1024] fp32
    const float* w = (const float*)d_in[1];   // [8192, 256] fp32
    float* out = (float*)d_out;

    // workspace layout: [wnh: K f32][wbf: K*D bf16][idx: N i32]  (~4.36 MB)
    char* ws = (char*)d_ws;
    float*          wnh  = (float*)ws;
    unsigned short* wbf  = (unsigned short*)(ws + (size_t)NKD_K * 4);
    int*            idxb = (int*)(ws + (size_t)NKD_K * 4 + (size_t)NKD_K * NKD_D * 2);

    vq_prep  <<<NKD_K, 256, 0, stream>>>(w, wnh, wbf, out);
    vq_argmin<<<NKD_N / 128, 256, 0, stream>>>(x, wbf, wnh, idxb, out);
    vq_out   <<<NKD_N / 128, 256, 0, stream>>>(x, w, idxb, out);
}
